// SelfAttention_20289425506739
// MI455X (gfx1250) — matrix operand order repaired and checked by reading, not compiled, not run on hardware
//
#include <hip/hip_runtime.h>

typedef __attribute__((ext_vector_type(16))) _Float16 v16h;
typedef __attribute__((ext_vector_type(8)))  _Float16 v8h;
typedef __attribute__((ext_vector_type(8)))  float    v8f;

#define DIM 1536
#define NH  16
#define HD  96
#define BB  2
#define SS  2048
#define MT  (BB*SS)          // 4096 rows total

// ---------------------------------------------------------------------------
// f32 -> f16 elementwise convert
// ---------------------------------------------------------------------------
__global__ __launch_bounds__(256)
void cvt_f32_f16(const float* __restrict__ src, _Float16* __restrict__ dst, int n) {
    int i = blockIdx.x * blockDim.x + threadIdx.x;
    if (i < n) dst[i] = (_Float16)src[i];
}

// ---------------------------------------------------------------------------
// GEMM: D[M,N] = A[M,K] @ Bw[N,K]^T   (A,Bw f16 row-major; D f16 or f32)
// One wave -> 32x64 tile (two 16-row subtiles sharing each B fragment).
// Block = 128 threads (4 waves) -> 32 rows x 256 cols.
// ---------------------------------------------------------------------------
template <bool STORE_F32>
__global__ __launch_bounds__(128)
void gemm_wmma(const _Float16* __restrict__ A,
               const _Float16* __restrict__ Bw,
               void* __restrict__ D,
               int M, int N, int K) {
    const int lane = threadIdx.x & 31;
    const int wave = threadIdx.x >> 5;
    const int g    = lane >> 4;         // half-wave group
    const int l    = lane & 15;
    const int m0   = blockIdx.x * 32;
    const int n0   = blockIdx.y * 256 + wave * 64;

    v8f acc[2][4] = {};
    const _Float16* arow0 = A + (size_t)(m0 + l) * K;
    const _Float16* arow1 = A + (size_t)(m0 + 16 + l) * K;

    for (int k0 = 0; k0 < K; k0 += 32) {
        // prefetch next K-step (speculative, streams through L2)
        if (k0 + 32 < K) {
            __builtin_prefetch(arow0 + k0 + 32, 0, 0);
            __builtin_prefetch(arow1 + k0 + 32, 0, 0);
            __builtin_prefetch(Bw + (size_t)(n0 + l) * K + k0 + 32, 0, 0);
        }

        // A fragments: lanes 0-15 -> K {0..15}; lanes 16-31 -> K {16..31}
        v16h a0, a1;
        {
            v8h lo = *(const v8h*)(arow0 + k0 + (g ? 16 : 0));
            v8h hi = *(const v8h*)(arow0 + k0 + (g ? 24 : 8));
#pragma unroll
            for (int i = 0; i < 8; ++i) { a0[i] = lo[i]; a0[i + 8] = hi[i]; }
        }
        {
            v8h lo = *(const v8h*)(arow1 + k0 + (g ? 16 : 0));
            v8h hi = *(const v8h*)(arow1 + k0 + (g ? 24 : 8));
#pragma unroll
            for (int i = 0; i < 8; ++i) { a1[i] = lo[i]; a1[i + 8] = hi[i]; }
        }

#pragma unroll
        for (int t = 0; t < 4; ++t) {
            // B fragment: lane holds N = n0+t*16+l, K contiguous (g selects K half)
            const _Float16* bp = Bw + (size_t)(n0 + t * 16 + l) * K + k0 + g * 16;
            v16h b = *(const v16h*)bp;
            acc[0][t] = __builtin_amdgcn_wmma_f32_16x16x32_f16(
                false, a0, false, b, (short)0, acc[0][t], false, false);
            acc[1][t] = __builtin_amdgcn_wmma_f32_16x16x32_f16(
                false, a1, false, b, (short)0, acc[1][t], false, false);
        }
    }

#pragma unroll
    for (int u = 0; u < 2; ++u) {
#pragma unroll
        for (int t = 0; t < 4; ++t) {
            const int n = n0 + t * 16 + l;
#pragma unroll
            for (int r = 0; r < 8; ++r) {
                const int m = m0 + u * 16 + r + 8 * g;
                if (STORE_F32) ((float*)D)[(size_t)m * N + n] = acc[u][t][r];
                else ((_Float16*)D)[(size_t)m * N + n] = (_Float16)acc[u][t][r];
            }
        }
    }
}

// ---------------------------------------------------------------------------
// 3D RoPE applied in-place to q,k (f16). One thread per rotation pair.
// head dim 96 = 3 axes x 32; pair (i, i+16) within each 32-block.
// ---------------------------------------------------------------------------
__global__ __launch_bounds__(256)
void rope_qk(_Float16* __restrict__ q, _Float16* __restrict__ k,
             const int* __restrict__ xyz) {
    int t = blockIdx.x * blockDim.x + threadIdx.x;
    if (t >= BB * SS * NH * 3 * 16) return;
    const int pair = t & 15;  int u = t >> 4;
    const int axis = u % 3;   u /= 3;
    const int head = u & 15;  u >>= 4;
    const int s    = u & (SS - 1);
    const int b    = u >> 11;

    const int pos = xyz[((size_t)(b * SS + s)) * 3 + axis];
    // inv_freq = 10000^(-pair/16) ; ln(10000)=9.210340371976184
    const float inv = __expf(-(float)pair * (9.210340371976184f / 16.0f));
    const float ang = (float)pos * inv;
    const float c = __cosf(ang), sn = __sinf(ang);

    const size_t base = ((size_t)(b * SS + s)) * DIM + head * HD + axis * 32 + pair;
    float q1 = (float)q[base], q2 = (float)q[base + 16];
    q[base]      = (_Float16)(q1 * c - q2 * sn);
    q[base + 16] = (_Float16)(q2 * c + q1 * sn);
    float k1 = (float)k[base], k2 = (float)k[base + 16];
    k[base]      = (_Float16)(k1 * c - k2 * sn);
    k[base + 16] = (_Float16)(k2 * c + k1 * sn);
}

// ---------------------------------------------------------------------------
// V transpose: v[b][s][h*HD+d] -> vt[b][h][d][s]  (f16)
// ---------------------------------------------------------------------------
__global__ __launch_bounds__(256)
void transpose_v(const _Float16* __restrict__ v, _Float16* __restrict__ vt) {
    int t = blockIdx.x * blockDim.x + threadIdx.x;
    if (t >= MT * DIM) return;
    const int n = t % DIM;
    const int s = (t / DIM) % SS;
    const int b = t / (DIM * SS);
    const int head = n / HD, d = n % HD;
    vt[(((size_t)(b * NH + head)) * HD + d) * SS + s] = v[t];
}

// ---------------------------------------------------------------------------
// Flash attention: one wave handles (b, head, 16-row Q tile), full S sweep.
// ---------------------------------------------------------------------------
__global__ __launch_bounds__(128)
void attn_kernel(const _Float16* __restrict__ qh,
                 const _Float16* __restrict__ kh,
                 const _Float16* __restrict__ vt,
                 _Float16* __restrict__ ao) {
    __shared__ _Float16 plds[4][16][32];   // per-wave P tile (C-layout -> A-layout)

    const int lane = threadIdx.x & 31;
    const int wave = threadIdx.x >> 5;
    const int g    = lane >> 4;
    const int l    = lane & 15;

    const int task = blockIdx.x * 4 + wave;          // BB*NH*(SS/16) = 4096 tasks
    const int qt   = task & 127;
    const int head = (task >> 7) & 15;
    const int b    = task >> 11;
    const int i0   = qt * 16;
    const float scale = 0.10206207261596575f;        // 1/sqrt(96)

    // Q A-fragments for K = 96 (3 fragments of 16x32), loaded once
    v16h aq[3];
    const _Float16* qrow = qh + ((size_t)(b * SS + i0 + l)) * DIM + head * HD;
#pragma unroll
    for (int ks = 0; ks < 3; ++ks) {
        v8h lo = *(const v8h*)(qrow + ks * 32 + (g ? 16 : 0));
        v8h hi = *(const v8h*)(qrow + ks * 32 + (g ? 24 : 8));
#pragma unroll
        for (int i = 0; i < 8; ++i) { aq[ks][i] = lo[i]; aq[ks][i + 8] = hi[i]; }
    }

    v8f o[6] = {};
    float mrow[8], lrow[8];
#pragma unroll
    for (int r = 0; r < 8; ++r) { mrow[r] = -1e30f; lrow[r] = 0.0f; }

    const _Float16* kbase = kh + ((size_t)b * SS) * DIM + head * HD;
    const _Float16* vtb   = vt + ((size_t)(b * NH + head)) * HD * SS;

    for (int j0 = 0; j0 < SS; j0 += 32) {
        // ---- S = Q @ K^T for 32 columns (two 16x16 tiles) ----
        v8f s0 = {}, s1 = {};
#pragma unroll
        for (int ks = 0; ks < 3; ++ks) {
            v16h b0 = *(const v16h*)(kbase + (size_t)(j0 + l) * DIM + ks * 32 + g * 16);
            s0 = __builtin_amdgcn_wmma_f32_16x16x32_f16(
                false, aq[ks], false, b0, (short)0, s0, false, false);
            v16h b1 = *(const v16h*)(kbase + (size_t)(j0 + 16 + l) * DIM + ks * 32 + g * 16);
            s1 = __builtin_amdgcn_wmma_f32_16x16x32_f16(
                false, aq[ks], false, b1, (short)0, s1, false, false);
        }

        // ---- online softmax over the 32 new columns ----
        float p0[8], p1[8], alpha[8];
#pragma unroll
        for (int r = 0; r < 8; ++r) {
            float v0 = s0[r] * scale, v1 = s1[r] * scale;
            float mx = fmaxf(v0, v1);
#pragma unroll
            for (int off = 1; off < 16; off <<= 1)
                mx = fmaxf(mx, __shfl_xor(mx, off, 32));
            const float mn = fmaxf(mrow[r], mx);
            const float al = __expf(mrow[r] - mn);
            const float e0 = __expf(v0 - mn), e1 = __expf(v1 - mn);
            float sum = e0 + e1;
#pragma unroll
            for (int off = 1; off < 16; off <<= 1)
                sum += __shfl_xor(sum, off, 32);
            lrow[r] = lrow[r] * al + sum;
            mrow[r] = mn;
            p0[r] = e0; p1[r] = e1; alpha[r] = al;
        }
#pragma unroll
        for (int t = 0; t < 6; ++t)
#pragma unroll
            for (int r = 0; r < 8; ++r) o[t][r] *= alpha[r];

        // ---- reshape P: C-layout -> LDS -> A-layout (same wave, DS in-order) ----
#pragma unroll
        for (int r = 0; r < 8; ++r) {
            plds[wave][r + 8 * g][l]      = (_Float16)p0[r];
            plds[wave][r + 8 * g][16 + l] = (_Float16)p1[r];
        }
        v16h pa;
        {
            const _Float16* pr = &plds[wave][l][0];
            v8h lo = *(const v8h*)(pr + (g ? 16 : 0));
            v8h hi = *(const v8h*)(pr + (g ? 24 : 8));
#pragma unroll
            for (int i = 0; i < 8; ++i) { pa[i] = lo[i]; pa[i + 8] = hi[i]; }
        }

        // ---- O += P @ V  (B operand from transposed V, K = sequence dim) ----
#pragma unroll
        for (int t = 0; t < 6; ++t) {
            v16h bv = *(const v16h*)(vtb + (size_t)(t * 16 + l) * SS + j0 + g * 16);
            o[t] = __builtin_amdgcn_wmma_f32_16x16x32_f16(
                false, pa, false, bv, (short)0, o[t], false, false);
        }
    }

    // ---- epilogue: normalize and store attn output (f16) ----
    _Float16* orow = ao + ((size_t)(b * SS + i0)) * DIM + head * HD;
#pragma unroll
    for (int r = 0; r < 8; ++r) {
        const float inv = 1.0f / lrow[r];
        const int m = r + 8 * g;
#pragma unroll
        for (int t = 0; t < 6; ++t)
            orow[(size_t)m * DIM + t * 16 + l] = (_Float16)(o[t][r] * inv);
    }
}

// ---------------------------------------------------------------------------
// Host-side launcher
// ---------------------------------------------------------------------------
extern "C" void kernel_launch(void* const* d_in, const int* in_sizes, int n_in,
                              void* d_out, int out_size, void* d_ws, size_t ws_size,
                              hipStream_t stream) {
    const int*   xyz = (const int*)  d_in[0];
    const float* h   = (const float*)d_in[1];
    const float* wq  = (const float*)d_in[2];
    const float* wk  = (const float*)d_in[3];
    const float* wv  = (const float*)d_in[4];
    const float* wo  = (const float*)d_in[5];
    float* out = (float*)d_out;

    char* ws = (char*)d_ws;
    size_t off = 0;
    auto carve = [&](size_t elems) -> _Float16* {
        _Float16* p = (_Float16*)(ws + off);
        off = (off + elems * sizeof(_Float16) + 255) & ~(size_t)255;
        return p;
    };
    const size_t WELEMS = (size_t)DIM * DIM;   // 2.36M
    const size_t XELEMS = (size_t)MT * DIM;    // 6.29M

    _Float16* wq_h = carve(WELEMS);
    _Float16* wk_h = carve(WELEMS);
    _Float16* wv_h = carve(WELEMS);
    _Float16* wo_h = carve(WELEMS);
    _Float16* h_h  = carve(XELEMS);
    _Float16* q_h  = carve(XELEMS);
    _Float16* k_h  = carve(XELEMS);
    _Float16* v_h  = carve(XELEMS);
    _Float16* vt_h = carve(XELEMS);
    _Float16* ao_h = carve(XELEMS);

    // 1) convert inputs to f16
    {
        int nw = (int)WELEMS, nx = (int)XELEMS;
        cvt_f32_f16<<<(nw + 255) / 256, 256, 0, stream>>>(wq, wq_h, nw);
        cvt_f32_f16<<<(nw + 255) / 256, 256, 0, stream>>>(wk, wk_h, nw);
        cvt_f32_f16<<<(nw + 255) / 256, 256, 0, stream>>>(wv, wv_h, nw);
        cvt_f32_f16<<<(nw + 255) / 256, 256, 0, stream>>>(wo, wo_h, nw);
        cvt_f32_f16<<<(nx + 255) / 256, 256, 0, stream>>>(h,  h_h,  nx);
    }

    // 2) QKV projections: q/k/v = h @ w^T   (32-row tiles per wave)
    dim3 ggrid(MT / 32, DIM / 256);
    gemm_wmma<false><<<ggrid, 128, 0, stream>>>(h_h, wq_h, q_h, MT, DIM, DIM);
    gemm_wmma<false><<<ggrid, 128, 0, stream>>>(h_h, wk_h, k_h, MT, DIM, DIM);
    gemm_wmma<false><<<ggrid, 128, 0, stream>>>(h_h, wv_h, v_h, MT, DIM, DIM);

    // 3) RoPE on q,k (in place) + transpose v
    {
        int nr = BB * SS * NH * 3 * 16;
        rope_qk<<<(nr + 255) / 256, 256, 0, stream>>>(q_h, k_h, xyz);
        int nt = (int)XELEMS;
        transpose_v<<<(nt + 255) / 256, 256, 0, stream>>>(v_h, vt_h);
    }

    // 4) attention (flash-style, 16-row Q tiles per wave)
    attn_kernel<<<(BB * NH * (SS / 16)) / 4, 128, 0, stream>>>(q_h, k_h, vt_h, ao_h);

    // 5) output projection: out = ao @ wo^T (f32 store)
    gemm_wmma<true><<<ggrid, 128, 0, stream>>>(ao_h, wo_h, out, MT, DIM, DIM);
}
